// MultiHeadAttentionLayer_75170517615192
// MI455X (gfx1250) — matrix operand order repaired and checked by reading, not compiled, not run on hardware
//
#include <hip/hip_runtime.h>

typedef __attribute__((ext_vector_type(16))) __bf16 v16bf;
typedef __attribute__((ext_vector_type(8)))  __bf16 v8bf;
typedef __attribute__((ext_vector_type(8)))  float  v8f;

#define B_  4
#define S_  1024
#define D_  1024
#define H_  16
#define HD_ 64

// Pack two f32 -> one dword of two bf16 (lowered to v_cvt_pk_bf16_f32)
static __device__ __forceinline__ unsigned pack_bf16(float x, float y) {
    union { __bf16 h[2]; unsigned u; } p;
    p.h[0] = (__bf16)x; p.h[1] = (__bf16)y;
    return p.u;
}

// A-fragment (16x32 bf16) from a row-major row: af[0..15]=row[2kb..2kb+15]
// (contiguous k order). Two 16B LDS loads.
static __device__ __forceinline__ v16bf frag_row(const __bf16* base, int kb) {
    v8bf lo = *(const v8bf*)(base + 2 * kb);
    v8bf hi = *(const v8bf*)(base + 2 * kb + 8);
    return __builtin_shufflevector(lo, hi, 0,1,2,3,4,5,6,7,8,9,10,11,12,13,14,15);
}

// Fragment whose 16 bf16 are contiguous in LDS (32 bytes).
static __device__ __forceinline__ v16bf frag_contig(const __bf16* p) {
    v8bf lo = *(const v8bf*)(p);
    v8bf hi = *(const v8bf*)(p + 8);
    return __builtin_shufflevector(lo, hi, 0,1,2,3,4,5,6,7,8,9,10,11,12,13,14,15);
}

// ---------------------------------------------------------------------------
// Projection GEMM: C[M x 1024] = A[M x 1024] @ W[1024 x 1024] + bias
//   mode 0: f32 row-major [M, D] to outF        (final output projection)
//   mode 1: bf16 head-major [B,H,S,HD] to outBF (Q/K/V projections)
// Tile 128(M) x 64(N), K-chunk 32, 8 waves; wave owns one 16-row M tile and
// all four N tiles. W tile staged pair-swizzled so B-fragments are contiguous.
// All fragment loads for a K-chunk are issued before the WMMA chain so the
// 10 ds_load_b128 form one clause behind a single s_wait_dscnt.
// ---------------------------------------------------------------------------
#define TM 128
#define TN 64
#define KC 32

__global__ __launch_bounds__(256)
void proj_gemm_kernel(const float* __restrict__ A,
                      const float* __restrict__ W,
                      const float* __restrict__ bias,
                      float* __restrict__ outF,
                      __bf16* __restrict__ outBF,
                      int mode)
{
    __shared__ __bf16   sA[TM][KC];      // 8 KB, row-major
    __shared__ unsigned sBfrag[TN][16];  // 4 KB, pair-swizzled: [n][k/2]

    const int tid  = threadIdx.x;
    const int lane = tid & 31;
    const int wave = tid >> 5;
    const int half = lane >> 4;
    const int nl   = lane & 15;

    const int m0 = blockIdx.x * TM;
    const int n0 = blockIdx.y * TN;

    v8f acc[4];
    for (int t = 0; t < 4; ++t)
        for (int j = 0; j < 8; ++j) acc[t][j] = 0.0f;

    for (int kc = 0; kc < D_; kc += KC) {
        __syncthreads();  // protect LDS reuse from previous iteration's reads

        // Stage A chunk (128x32 f32 -> bf16, row-major): 1024 float4, 4/thread,
        // each packed to one 8B LDS store.
        for (int i = 0; i < 4; ++i) {
            int idx = tid + i * 256;
            int row = idx >> 3;          // 8 float4 per 32-float row
            int c4  = idx & 7;
            float4 f = *(const float4*)(A + (size_t)(m0 + row) * D_ + kc + c4 * 4);
            uint2 pk;
            pk.x = pack_bf16(f.x, f.y);
            pk.y = pack_bf16(f.z, f.w);
            *(uint2*)&sA[row][c4 * 4] = pk;
        }
        // Stage W chunk (32x64) pair-swizzled: thread (p = K-pair, nq = col/4)
        // reads two rows, packs 4 dwords of (W[2p][n], W[2p+1][n]).
        {
            int p  = tid & 15;
            int nq = tid >> 4;
            float4 fa = *(const float4*)(W + (size_t)(kc + 2*p)     * D_ + n0 + nq * 4);
            float4 fb = *(const float4*)(W + (size_t)(kc + 2*p + 1) * D_ + n0 + nq * 4);
            sBfrag[nq*4 + 0][p] = pack_bf16(fa.x, fb.x);
            sBfrag[nq*4 + 1][p] = pack_bf16(fa.y, fb.y);
            sBfrag[nq*4 + 2][p] = pack_bf16(fa.z, fb.z);
            sBfrag[nq*4 + 3][p] = pack_bf16(fa.w, fb.w);
        }
        // Prefetch next A chunk (global_prefetch_b8)
        if (kc + KC < D_)
            __builtin_prefetch(A + (size_t)(m0 + (tid >> 1)) * D_ + kc + KC, 0, 1);
        __syncthreads();

        // Batch ALL fragment loads, then run the WMMA chain.
        v16bf af = frag_row(&sA[wave * 16 + nl][0], half * 8);
        v16bf bfr[4];
#pragma unroll
        for (int nt = 0; nt < 4; ++nt)
            bfr[nt] = frag_contig((const __bf16*)&sBfrag[nt * 16 + nl][half * 8]);
#pragma unroll
        for (int nt = 0; nt < 4; ++nt)
            acc[nt] = __builtin_amdgcn_wmma_f32_16x16x32_bf16(
                false, af, false, bfr[nt], (short)0, acc[nt], false, false);
    }

    // Epilogue: C layout VGPR j -> M = half*8 + j, N = lane & 15
    for (int nt = 0; nt < 4; ++nt) {
        for (int j = 0; j < 8; ++j) {
            int m = m0 + wave * 16 + half * 8 + j;
            int n = n0 + nt * 16 + nl;
            float vv = acc[nt][j] + bias[n];
            if (mode == 0) {
                outF[(size_t)m * D_ + n] = vv;
            } else {
                int b = m >> 10, s = m & (S_ - 1);
                int h = n >> 6,  d = n & (HD_ - 1);
                outBF[(((size_t)(b * H_ + h) * S_) + s) * HD_ + d] = (__bf16)vv;
            }
        }
    }
}

// ---------------------------------------------------------------------------
// Attention: one workgroup per (b*H+h, 16-query-row tile).
// Phase 1: QK^T/sqrt(HD) + mask -> 16x1024 score panel in LDS (WMMA).
//          Q fragments hoisted out of the key-tile sweep (loop-invariant).
// Phase 2: wave-parallel softmax; probs f32 -> d_out, bf16 -> LDS.
// Phase 3: P @ V via WMMA; V chunk staged pair-swizzled for contiguous B-frags.
// ---------------------------------------------------------------------------
__global__ __launch_bounds__(256)
void attn_kernel(const __bf16* __restrict__ qbf,
                 const __bf16* __restrict__ kbf,
                 const __bf16* __restrict__ vbf,
                 const int*    __restrict__ mask,
                 float* __restrict__ attn_out,
                 float* __restrict__ xctx)
{
    __shared__ float    sScore[16][S_];    // 64 KB
    __shared__ __bf16   sProb[16][S_];     // 32 KB
    __shared__ __bf16   sK[8][16][HD_];    // 16 KB: per-wave K tiles (row-major)
    __shared__ unsigned sVfrag[HD_][16];   //  4 KB: pair-swizzled V chunk [d][k/2]
    __shared__ __bf16   sQ[16][HD_];       //  2 KB
    __shared__ int      sMask[S_];         //  4 KB

    const int tid  = threadIdx.x;
    const int lane = tid & 31;
    const int wave = tid >> 5;
    const int half = lane >> 4;
    const int nl   = lane & 15;

    const int bh = blockIdx.x;             // b*H + h
    const int qt = blockIdx.y;             // query tile of 16 rows
    const int b  = bh >> 4;
    const size_t headBase = (size_t)bh * S_ * HD_;

    // Load Q tile (straight 2KB copy, 128 uint4) and key mask
    if (tid < 128)
        ((uint4*)&sQ[0][0])[tid] =
            ((const uint4*)(qbf + headBase + (size_t)qt * 16 * HD_))[tid];
    for (int i = tid; i < S_; i += 256) sMask[i] = mask[b * S_ + i];
    __syncthreads();

    // ---- Phase 1: scores. Each wave sweeps 8 key tiles of 16 keys. ----
    // Q fragments are invariant across the sweep: load once.
    const v16bf aq0 = frag_row(&sQ[nl][0],  half * 8);
    const v16bf aq1 = frag_row(&sQ[nl][32], half * 8);

    for (int it = 0; it < 8; ++it) {
        int kt = wave * 8 + it;
        {   // wave-private K-tile staging (DS ops are in-order per wave)
            const uint4* src = (const uint4*)(kbf + headBase + (size_t)kt * 16 * HD_);
            uint4* dst = (uint4*)&sK[wave][0][0];
            for (int i = 0; i < 4; ++i) dst[lane + i * 32] = src[lane + i * 32];
        }
        // Both B fragments (B(k,n) = K[n][k], contiguous along d), then 2 WMMAs.
        v16bf bk0 = frag_contig(&sK[wave][nl][ 0 + half * 16]);
        v16bf bk1 = frag_contig(&sK[wave][nl][32 + half * 16]);
        v8f e;
        for (int j = 0; j < 8; ++j) e[j] = 0.0f;
        e = __builtin_amdgcn_wmma_f32_16x16x32_bf16(
            false, aq0, false, bk0, (short)0, e, false, false);
        e = __builtin_amdgcn_wmma_f32_16x16x32_bf16(
            false, aq1, false, bk1, (short)0, e, false, false);

        for (int j = 0; j < 8; ++j) {
            int qr  = half * 8 + j;
            int key = kt * 16 + nl;
            float sc = e[j] * 0.125f;                     // 1/sqrt(64)
            sScore[qr][key] = (sMask[key] == 0) ? -1.0e10f : sc;
        }
    }
    __syncthreads();

    // ---- Phase 2: softmax, 2 rows per wave, lanes stride the 1024 keys ----
    for (int rr = 0; rr < 2; ++rr) {
        int row = wave * 2 + rr;
        float mx = -3.0e38f;
        for (int i = lane; i < S_; i += 32) mx = fmaxf(mx, sScore[row][i]);
        for (int off = 16; off > 0; off >>= 1) mx = fmaxf(mx, __shfl_xor(mx, off, 32));
        float sum = 0.0f;
        for (int i = lane; i < S_; i += 32) {
            float p = __expf(sScore[row][i] - mx);
            sScore[row][i] = p;
            sum += p;
        }
        for (int off = 16; off > 0; off >>= 1) sum += __shfl_xor(sum, off, 32);
        float inv = 1.0f / sum;
        size_t obase = ((size_t)bh * S_ + qt * 16 + row) * S_;
        for (int i = lane; i < S_; i += 32) {
            float pn = sScore[row][i] * inv;
            attn_out[obase + i] = pn;                     // [B,H,S,S] output
            sProb[row][i] = (__bf16)pn;
        }
    }
    __syncthreads();

    // ---- Phase 3: context = P @ V. Waves 0..3 own 16-wide d tiles. ----
    v8f ctx;
    for (int j = 0; j < 8; ++j) ctx[j] = 0.0f;

    for (int kc = 0; kc < S_; kc += 32) {
        __syncthreads();
        {   // stage V[32 keys][64 d] pair-swizzled: thread (p = key-pair, dq = d/4)
            int p  = tid & 15;
            int dq = tid >> 4;
            union { uint2 v; __bf16 h[4]; } ra, rb;
            ra.v = *(const uint2*)(vbf + headBase + (size_t)(kc + 2*p)     * HD_ + dq * 4);
            rb.v = *(const uint2*)(vbf + headBase + (size_t)(kc + 2*p + 1) * HD_ + dq * 4);
            for (int i = 0; i < 4; ++i) {
                union { __bf16 h[2]; unsigned u; } pk;
                pk.h[0] = ra.h[i]; pk.h[1] = rb.h[i];
                sVfrag[dq * 4 + i][p] = pk.u;
            }
        }
        __syncthreads();
        if (wave < 4) {                                   // wave-uniform: EXEC all-1s
            v16bf ap = frag_row(&sProb[nl][kc], half * 8);
            v16bf bv = frag_contig((const __bf16*)&sVfrag[wave * 16 + nl][half * 8]);
            ctx = __builtin_amdgcn_wmma_f32_16x16x32_bf16(
                false, ap, false, bv, (short)0, ctx, false, false);
        }
    }
    if (wave < 4) {
        int h = bh & 15;
        for (int j = 0; j < 8; ++j) {
            int qr = half * 8 + j;
            int s  = qt * 16 + qr;
            int d  = wave * 16 + nl;
            xctx[((size_t)b * S_ + s) * D_ + h * HD_ + d] = ctx[j];   // [B,S,D]
        }
    }
}

// ---------------------------------------------------------------------------
// Launch: Q/K/V projections (bf16 head-major) -> attention -> output proj.
// Workspace: qbf(8MB) | kbf(8MB) | vbf(8MB) | xctx(16MB) = 40MB.
// d_out: outputs [B,S,D] f32 (4M floats) then attention [B,H,S,S] f32 (64M).
// ---------------------------------------------------------------------------
extern "C" void kernel_launch(void* const* d_in, const int* in_sizes, int n_in,
                              void* d_out, int out_size, void* d_ws, size_t ws_size,
                              hipStream_t stream) {
    const float* q    = (const float*)d_in[0];
    const float* k    = (const float*)d_in[1];
    const float* v    = (const float*)d_in[2];
    const int*   mask = (const int*)  d_in[3];
    const float* Wq = (const float*)d_in[4];  const float* bq = (const float*)d_in[5];
    const float* Wk = (const float*)d_in[6];  const float* bk = (const float*)d_in[7];
    const float* Wv = (const float*)d_in[8];  const float* bv = (const float*)d_in[9];
    const float* Wo = (const float*)d_in[10]; const float* bo = (const float*)d_in[11];

    float* out  = (float*)d_out;                              // [B,S,D]
    float* attn = out + (size_t)B_ * S_ * D_;                 // [B,H,S,S]

    char* ws = (char*)d_ws;
    __bf16* qbf  = (__bf16*)(ws);                             // 8 MB
    __bf16* kbf  = (__bf16*)(ws + ((size_t)8  << 20));        // 8 MB
    __bf16* vbf  = (__bf16*)(ws + ((size_t)16 << 20));        // 8 MB
    float*  xctx = (float*) (ws + ((size_t)24 << 20));        // 16 MB

    const dim3 gG(B_ * S_ / TM, D_ / TN);                     // 32 x 16
    proj_gemm_kernel<<<gG, 256, 0, stream>>>(q, Wq, bq, nullptr, qbf, 1);
    proj_gemm_kernel<<<gG, 256, 0, stream>>>(k, Wk, bk, nullptr, kbf, 1);
    proj_gemm_kernel<<<gG, 256, 0, stream>>>(v, Wv, bv, nullptr, vbf, 1);

    attn_kernel<<<dim3(B_ * H_, S_ / 16), 256, 0, stream>>>(qbf, kbf, vbf,
                                                            mask, attn, xctx);

    proj_gemm_kernel<<<gG, 256, 0, stream>>>(xctx, Wo, bo, out, nullptr, 0);
}
